// DynGATModel_87763361727278
// MI455X (gfx1250) — matrix operand order repaired and checked by reading, not compiled, not run on hardware
//
#include <hip/hip_runtime.h>
#include <hip/hip_bf16.h>

// ---------------------------------------------------------------------------
// Dynamic 2-layer GAT over L=4 graphs for MI455X (gfx1250, wave32, WMMA).
//
// GEMMs (x@W1 once; h@W2 per graph) use v_wmma_f32_16x16x32_f16:
//   - B (weights) pre-swizzled once per call into the per-lane fragment layout
//     (lane%16 = column, K contiguous within lane, lane/16 selects K half),
//     so the B fragment is one contiguous 32-byte load per K-step.
//   - A fragment built with 8x float2 loads + f32->f16 convert per K-step,
//     in the same contiguous K order as B (element i = K 16*(lane/16) + i
//     within the 32-wide K-step).
// Edge softmax/aggregation uses ordered-uint atomicMax for segment max and
// native f32 atomicAdd for segment sum / scatter-aggregate; feat + accumulator
// (77 MB) sit in the 192 MB L2 so the random gathers/scatters stay on-chip.
// ---------------------------------------------------------------------------

typedef __attribute__((ext_vector_type(16))) _Float16 v16h;
typedef __attribute__((ext_vector_type(8)))  float    v8f;

#define GAT_N   50000
#define GAT_E   800000
#define GAT_L   4
#define GAT_INF 256
#define GAT_HF  64
#define GAT_H1  3
#define GAT_C1  (GAT_H1 * GAT_HF)   // 192

// ------------------------------- helpers -----------------------------------

__device__ __forceinline__ unsigned f2ord(float f) {
    unsigned b = __float_as_uint(f);
    return (b & 0x80000000u) ? ~b : (b | 0x80000000u);
}
__device__ __forceinline__ float ord2f(unsigned o) {
    unsigned b = (o & 0x80000000u) ? (o & 0x7FFFFFFFu) : ~o;
    return __uint_as_float(b);
}
__device__ __forceinline__ float lrelu(float v) { return v > 0.f ? v : 0.2f * v; }

// ------------------------- weight pre-swizzle ------------------------------
// Bsw element index = ((ct*(K/32) + ks)*32 + lane)*16 + i
// value = B[ ks*32 + 16*(lane>>4) + i ][ ct*16 + (lane&15) ]   (B is K x Nc)
__global__ void swizzleB_kernel(const float* __restrict__ B, _Float16* __restrict__ Bsw,
                                int K, int Nc) {
    int idx = blockIdx.x * blockDim.x + threadIdx.x;
    int total = K * Nc;
    if (idx >= total) return;
    int i    = idx & 15;
    int lane = (idx >> 4) & 31;
    int rest = idx >> 9;
    int nks  = K >> 5;
    int ks   = rest % nks;
    int ct   = rest / nks;
    int n    = (ct << 4) + (lane & 15);
    int k    = (ks << 5) + ((lane >> 4) << 4) + i;
    Bsw[idx] = (_Float16)B[(size_t)k * Nc + n];
}

// ------------------------------ WMMA GEMM ----------------------------------
// C[M x Nc] = A[M x K] @ B[K x Nc].  Block = 256 threads = 8 waves (4x2 tiles)
// covering 64 rows x 32 cols.  K, Nc multiples of 32/16.
__global__ void wmma_gemm_kernel(const float* __restrict__ A,
                                 const _Float16* __restrict__ Bsw,
                                 float* __restrict__ C,
                                 int M, int K, int Nc) {
    const int wid  = threadIdx.x >> 5;
    const int lane = threadIdx.x & 31;
    const int wrow = wid & 3;          // 4 wave-rows
    const int wcol = wid >> 2;         // 2 wave-cols
    const int half = lane >> 4;
    const int l16  = lane & 15;

    const int tileRow0 = blockIdx.x * 64 + wrow * 16;
    const int tileCol0 = blockIdx.y * 32 + wcol * 16;
    const int ct  = tileCol0 >> 4;
    const int nks = K >> 5;

    int m = tileRow0 + l16;
    if (m >= M) m = M - 1;                       // clamp (stores masked later)
    const float* arow = A + (size_t)m * K;
    const v16h* bbase = (const v16h*)Bsw + (size_t)ct * nks * 32;

    v8f acc = {};
    for (int ks = 0; ks < nks; ++ks) {
        const int k0 = ks << 5;
        v16h a;
#pragma unroll
        for (int p = 0; p < 8; ++p) {
            int kb = 2 * p + 16 * half;
            float2 f = *(const float2*)(arow + k0 + kb);
            a[2 * p]     = (_Float16)f.x;
            a[2 * p + 1] = (_Float16)f.y;
        }
        v16h b = bbase[(size_t)ks * 32 + lane];
        acc = __builtin_amdgcn_wmma_f32_16x16x32_f16(
                  false, a, false, b, (short)0, acc, false, false);
    }

#pragma unroll
    for (int v = 0; v < 8; ++v) {
        int row = tileRow0 + v + 8 * half;
        if (row < M) C[(size_t)row * Nc + tileCol0 + l16] = acc[v];
    }
}

// --------------------------- attention terms -------------------------------
// el[n,h] = dot(feat[n,h,:], al[h,:]);  er likewise.
__global__ void eler_kernel(const float* __restrict__ feat,
                            const float* __restrict__ al,
                            const float* __restrict__ ar,
                            float* __restrict__ el, float* __restrict__ er,
                            int Nn, int H, int D) {
    int idx = blockIdx.x * blockDim.x + threadIdx.x;
    if (idx >= Nn * H) return;
    int n = idx / H, h = idx - n * H;
    const float* f = feat + (size_t)n * H * D + (size_t)h * D;
    const float* a = al + h * D;
    const float* r = ar + h * D;
    float sl = 0.f, sr = 0.f;
    for (int d = 0; d < D; ++d) { float x = f[d]; sl += x * a[d]; sr += x * r[d]; }
    el[idx] = sl;
    er[idx] = sr;
}

// ------------------------------ edge kernels -------------------------------
__global__ void edge_max_kernel(const int* __restrict__ src, const int* __restrict__ dst,
                                const float* __restrict__ el, const float* __restrict__ er,
                                unsigned* __restrict__ m, int E, int H) {
    int idx = blockIdx.x * blockDim.x + threadIdx.x;
    if (idx >= E * H) return;
    int e = idx / H, h = idx - e * H;
    int s = src[e], d = dst[e];
    float v = lrelu(el[s * H + h] + er[d * H + h]);
    atomicMax(&m[d * H + h], f2ord(v));
}

__global__ void edge_exp_kernel(const int* __restrict__ src, const int* __restrict__ dst,
                                const float* __restrict__ el, const float* __restrict__ er,
                                const unsigned* __restrict__ m,
                                float* __restrict__ ssum, float* __restrict__ exb,
                                int E, int H) {
    int idx = blockIdx.x * blockDim.x + threadIdx.x;
    if (idx >= E * H) return;
    int e = idx / H, h = idx - e * H;
    int s = src[e], d = dst[e];
    float v  = lrelu(el[s * H + h] + er[d * H + h]);
    float ex = __expf(v - ord2f(m[d * H + h]));
    exb[idx] = ex;
    atomicAdd(&ssum[d * H + h], ex);
}

__global__ void alpha_norm_kernel(const int* __restrict__ dst,
                                  float* __restrict__ exb,
                                  const float* __restrict__ ssum, int E, int H) {
    int idx = blockIdx.x * blockDim.x + threadIdx.x;
    if (idx >= E * H) return;
    int e = idx / H, h = idx - e * H;
    exb[idx] = exb[idx] / ssum[dst[e] * H + h];
}

// blockDim.x == H*D; each block processes EPB edges.
__global__ void edge_agg_kernel(const int* __restrict__ src, const int* __restrict__ dst,
                                const float* __restrict__ alpha,
                                const float* __restrict__ feat,
                                float* __restrict__ out,
                                int E, int H, int D, int EPB) {
    int hd = threadIdx.x;
    int HD = H * D;
    int h  = hd / D;
    int e0 = blockIdx.x * EPB;
    for (int i = 0; i < EPB; ++i) {
        int e = e0 + i;
        if (e >= E) break;
        int s = src[e], d = dst[e];
        float a = alpha[e * H + h];
        atomicAdd(&out[(size_t)d * HD + hd], a * feat[(size_t)s * HD + hd]);
    }
}

// ----------------------------- elementwise ---------------------------------
__global__ void bias_relu_kernel(float* __restrict__ x, const float* __restrict__ b,
                                 int total, int C) {
    int idx = blockIdx.x * blockDim.x + threadIdx.x;
    if (idx >= total) return;
    float v = x[idx] + b[idx % C];
    x[idx] = v > 0.f ? v : 0.f;
}

__global__ void bias_out_kernel(const float* __restrict__ x, const float* __restrict__ b,
                                float* __restrict__ out, int total, int C) {
    int idx = blockIdx.x * blockDim.x + threadIdx.x;
    if (idx >= total) return;
    out[idx] = x[idx] + b[idx % C];
}

// ------------------------------- launcher ----------------------------------
extern "C" void kernel_launch(void* const* d_in, const int* in_sizes, int n_in,
                              void* d_out, int out_size, void* d_ws, size_t ws_size,
                              hipStream_t stream) {
    const float* in_feat = (const float*)d_in[0];   // N x 256
    const float* W1      = (const float*)d_in[1];   // 256 x 192
    const float* al1     = (const float*)d_in[2];   // 3 x 64
    const float* ar1     = (const float*)d_in[3];
    const float* b1      = (const float*)d_in[4];   // 192
    const float* W2      = (const float*)d_in[5];   // 192 x 64
    const float* al2     = (const float*)d_in[6];   // 1 x 64
    const float* ar2     = (const float*)d_in[7];
    const float* b2      = (const float*)d_in[8];   // 64
    const int*   srcs    = (const int*)d_in[9];     // L x E
    const int*   dsts    = (const int*)d_in[10];
    float*       out     = (float*)d_out;           // L x N x 64

    (void)in_sizes; (void)n_in; (void)out_size; (void)ws_size;

    // -------- workspace carve-up (bump allocator, 256B aligned) ------------
    size_t off = 0;
    char* base = (char*)d_ws;
    auto carve = [&](size_t bytes) -> char* {
        char* p = base + off;
        off += (bytes + 255) & ~(size_t)255;
        return p;
    };
    float*     feat1 = (float*)carve((size_t)GAT_N * GAT_C1 * 4);   // 38.4 MB
    float*     el1   = (float*)carve((size_t)GAT_N * GAT_H1 * 4);
    float*     er1   = (float*)carve((size_t)GAT_N * GAT_H1 * 4);
    unsigned*  m1    = (unsigned*)carve((size_t)GAT_N * GAT_H1 * 4);
    float*     ss1   = (float*)carve((size_t)GAT_N * GAT_H1 * 4);
    float*     ex1   = (float*)carve((size_t)GAT_E * GAT_H1 * 4);   // 9.6 MB
    float*     agg1  = (float*)carve((size_t)GAT_N * GAT_C1 * 4);   // 38.4 MB (doubles as h1)
    _Float16*  W1sw  = (_Float16*)carve((size_t)GAT_INF * GAT_C1 * 2);
    _Float16*  W2sw  = (_Float16*)carve((size_t)GAT_C1 * GAT_HF * 2);
    float*     feat2 = (float*)carve((size_t)GAT_N * GAT_HF * 4);   // 12.8 MB
    float*     el2   = (float*)carve((size_t)GAT_N * 4);
    float*     er2   = (float*)carve((size_t)GAT_N * 4);
    unsigned*  m2    = (unsigned*)carve((size_t)GAT_N * 4);
    float*     ss2   = (float*)carve((size_t)GAT_N * 4);
    float*     ex2   = (float*)carve((size_t)GAT_E * 4);            // 3.2 MB
    float*     agg2  = (float*)carve((size_t)GAT_N * GAT_HF * 4);   // 12.8 MB

    const int TB = 256;
    auto cdiv = [](int a, int b) { return (a + b - 1) / b; };

    // -------- weights -> WMMA fragment layout (f32 -> f16) -----------------
    swizzleB_kernel<<<cdiv(GAT_INF * GAT_C1, TB), TB, 0, stream>>>(W1, W1sw, GAT_INF, GAT_C1);
    swizzleB_kernel<<<cdiv(GAT_C1 * GAT_HF, TB), TB, 0, stream>>>(W2, W2sw, GAT_C1, GAT_HF);

    // -------- layer-1 projection (shared by all graphs) --------------------
    dim3 g1(cdiv(GAT_N, 64), GAT_C1 / 32);
    wmma_gemm_kernel<<<g1, TB, 0, stream>>>(in_feat, W1sw, feat1, GAT_N, GAT_INF, GAT_C1);
    eler_kernel<<<cdiv(GAT_N * GAT_H1, TB), TB, 0, stream>>>(
        feat1, al1, ar1, el1, er1, GAT_N, GAT_H1, GAT_HF);

    for (int g = 0; g < GAT_L; ++g) {
        const int* src = srcs + (size_t)g * GAT_E;
        const int* dst = dsts + (size_t)g * GAT_E;

        // ---------- layer 1: edge softmax + aggregate ----------
        hipMemsetAsync(m1,   0, (size_t)GAT_N * GAT_H1 * 4, stream);
        hipMemsetAsync(ss1,  0, (size_t)GAT_N * GAT_H1 * 4, stream);
        hipMemsetAsync(agg1, 0, (size_t)GAT_N * GAT_C1 * 4, stream);

        int eh1 = GAT_E * GAT_H1;
        edge_max_kernel<<<cdiv(eh1, TB), TB, 0, stream>>>(src, dst, el1, er1, m1, GAT_E, GAT_H1);
        edge_exp_kernel<<<cdiv(eh1, TB), TB, 0, stream>>>(src, dst, el1, er1, m1, ss1, ex1, GAT_E, GAT_H1);
        alpha_norm_kernel<<<cdiv(eh1, TB), TB, 0, stream>>>(dst, ex1, ss1, GAT_E, GAT_H1);
        edge_agg_kernel<<<cdiv(GAT_E, 8), GAT_C1, 0, stream>>>(
            src, dst, ex1, feat1, agg1, GAT_E, GAT_H1, GAT_HF, 8);

        // h1 = relu(agg1 + b1), in place
        bias_relu_kernel<<<cdiv(GAT_N * GAT_C1, TB), TB, 0, stream>>>(
            agg1, b1, GAT_N * GAT_C1, GAT_C1);

        // ---------- layer 2: projection + edge softmax + aggregate ----------
        dim3 g2(cdiv(GAT_N, 64), GAT_HF / 32);
        wmma_gemm_kernel<<<g2, TB, 0, stream>>>(agg1, W2sw, feat2, GAT_N, GAT_C1, GAT_HF);
        eler_kernel<<<cdiv(GAT_N, TB), TB, 0, stream>>>(
            feat2, al2, ar2, el2, er2, GAT_N, 1, GAT_HF);

        hipMemsetAsync(m2,   0, (size_t)GAT_N * 4, stream);
        hipMemsetAsync(ss2,  0, (size_t)GAT_N * 4, stream);
        hipMemsetAsync(agg2, 0, (size_t)GAT_N * GAT_HF * 4, stream);

        edge_max_kernel<<<cdiv(GAT_E, TB), TB, 0, stream>>>(src, dst, el2, er2, m2, GAT_E, 1);
        edge_exp_kernel<<<cdiv(GAT_E, TB), TB, 0, stream>>>(src, dst, el2, er2, m2, ss2, ex2, GAT_E, 1);
        alpha_norm_kernel<<<cdiv(GAT_E, TB), TB, 0, stream>>>(dst, ex2, ss2, GAT_E, 1);
        edge_agg_kernel<<<cdiv(GAT_E, 8), GAT_HF, 0, stream>>>(
            src, dst, ex2, feat2, agg2, GAT_E, 1, GAT_HF, 8);

        // out[g] = agg2 + b2
        bias_out_kernel<<<cdiv(GAT_N * GAT_HF, TB), TB, 0, stream>>>(
            agg2, b2, out + (size_t)g * GAT_N * GAT_HF, GAT_N * GAT_HF, GAT_HF);
    }
}
